// MPNN_85426899517544
// MI455X (gfx1250) — matrix operand order repaired and checked by reading, not compiled, not run on hardware
//
#include <hip/hip_runtime.h>
#include <hip/hip_bf16.h>
#include <math.h>

// ---------------- problem constants (match reference) ----------------
#define NUMATOM   128
#define NELE      16
#define NDET      32
#define NWAVE     8
#define NCONTRACT 128
#define NGAUSS    8
#define RMAXL     2
#define NSPH      4
#define NPAIR     (NUMATOM*(NUMATOM-1))     // 16256
#define DEMB      16
#define DSPIN     4
#define FEAT_EMB  64
#define FEAT_MP   256
#define NLCH      5
#define NPATH     16
#define EMB_OUT   (2*NWAVE + RMAXL*NCONTRACT)   // 272
#define ENP       (NELE*(NUMATOM-NELE))         // 1792
#define NNP       ((NUMATOM-NELE)*(NUMATOM-NELE-1)) // 12432
#define EPSON     1e-10f

typedef __attribute__((ext_vector_type(16))) _Float16 v16h;
typedef __attribute__((ext_vector_type(8)))  float    v8f;

// CG path tables from _build_cg() (deterministic): (i1, i2, add, channel, sign)
#define CG_TABLES                                                            \
  constexpr int   I1[NPATH]={0,0,0,0, 1,2,3, 1,2,3, 1,2,2,3,3,1};            \
  constexpr int   I2[NPATH]={0,1,2,3, 0,0,0, 1,2,3, 2,1,3,2,1,3};            \
  constexpr int   IA[NPATH]={0,1,2,3, 1,2,3, 0,0,0, 3,3,1,1,2,2};            \
  constexpr int   IC[NPATH]={0,1,1,1, 2,2,2, 3,3,3, 4,4,4,4,4,4};            \
  constexpr float SGN[NPATH]={1,1,1,1, 1,1,1, 1,1,1, 1,-1,1,-1,1,-1};

// Async copy of one 16B chunk (global -> LDS), ASYNCcnt-tracked (CDNA5 TDM-lite path).
__device__ __forceinline__ void async_g2l_b128(const void* gptr, void* lptr) {
  uint32_t l = (uint32_t)(uintptr_t)lptr;          // LDS address: low 32 bits of the pointer
  uint64_t g = (uint64_t)(uintptr_t)gptr;
  asm volatile("global_load_async_to_lds_b128 %0, %1, off" :: "v"(l), "v"(g) : "memory");
}
__device__ __forceinline__ void async_wait0() {
  asm volatile("s_wait_asynccnt 0x0" ::: "memory");
}

// ---------------- f32 (KxN) -> f16 transposed (NxK) weight conversion ----------------
__global__ void k_cvt_tr(const float* __restrict__ s, _Float16* __restrict__ d,
                         int K, int N) {
  int i = blockIdx.x * 256 + threadIdx.x;
  if (i >= K * N) return;
  int k = i / N, n = i % N;
  d[(size_t)n * K + k] = (_Float16)s[i];
}

// ---------------- WMMA GEMM: C = epi(A[M,K] @ W[K,N] + bias), SiLU / residual fused ----
// One wave computes a 16x32 strip (two v_wmma_f32_16x16x32_f16 per K-step).
// A: f32 (lda stride, optional row gather), M must be a multiple of 16 (true for all
// call sites here). Wt: f16 *transposed* N x K so a B fragment is one contiguous v16h.
__global__ __launch_bounds__(32)
void k_gemm_wmma(const float* __restrict__ A, int lda,
                 const int* __restrict__ rowidx,
                 const _Float16* __restrict__ Wt,
                 const float* __restrict__ bias,
                 const float* __restrict__ Res,
                 float* __restrict__ C,
                 int M, int K, int N, int act)
{
  const int lane  = threadIdx.x;            // 0..31
  const int g     = lane >> 4;              // lane half
  const int lm    = lane & 15;
  const int mbase = blockIdx.y * 16;
  const int nbase = blockIdx.x * 32;
  const int arow  = mbase + lm;                         // < M always (M % 16 == 0)
  const int asrc  = rowidx ? rowidx[arow] : arow;
  const float* Arow = A + (size_t)asrc * lda;
  const int n0 = nbase + lm, n1 = n0 + 16;
  const _Float16* Wr0 = Wt + (size_t)(n0 < N ? n0 : N - 1) * K;  // clamped row addr
  const _Float16* Wr1 = Wt + (size_t)(n1 < N ? n1 : N - 1) * K;

  v8f c0 = {}, c1 = {};
  const int Km = K & ~31;
  int k0 = 0;
  // ---- guard-free main loop: 8x b128 loads + 16 cvt + 2 wmma per step ----
  for (; k0 < Km; k0 += 32) {
    if (k0 + 32 < Km) {
      __builtin_prefetch(Arow + k0 + 32, 0, 1);        // global_prefetch
      __builtin_prefetch(Wr0 + k0 + 32, 0, 1);
      __builtin_prefetch(Wr1 + k0 + 32, 0, 1);
    }
    const float* ap = Arow + k0 + (g << 4);
    float4 fa = *(const float4*)(ap + 0);              // A: lanes g=0 K 0-15 / g=1 K 16-31
    float4 fb = *(const float4*)(ap + 4);
    float4 fc = *(const float4*)(ap + 8);              //    (contiguous, matches B)
    float4 fd = *(const float4*)(ap + 12);
    v16h a;
    a[0]=(_Float16)fa.x;  a[1]=(_Float16)fa.y;  a[2]=(_Float16)fa.z;  a[3]=(_Float16)fa.w;
    a[4]=(_Float16)fb.x;  a[5]=(_Float16)fb.y;  a[6]=(_Float16)fb.z;  a[7]=(_Float16)fb.w;
    a[8]=(_Float16)fc.x;  a[9]=(_Float16)fc.y;  a[10]=(_Float16)fc.z; a[11]=(_Float16)fc.w;
    a[12]=(_Float16)fd.x; a[13]=(_Float16)fd.y; a[14]=(_Float16)fd.z; a[15]=(_Float16)fd.w;
    v16h b0 = *(const v16h*)(Wr0 + k0 + (g << 4));     // B: contiguous 16 halves
    v16h b1 = *(const v16h*)(Wr1 + k0 + (g << 4));
    c0 = __builtin_amdgcn_wmma_f32_16x16x32_f16(false, a, false, b0, (short)0, c0, false, false);
    c1 = __builtin_amdgcn_wmma_f32_16x16x32_f16(false, a, false, b1, (short)0, c1, false, false);
  }
  // ---- K tail (K = 4 / 16 layers): branch-free clamped loads + cndmask zeroing ----
  if (k0 < K) {
    v16h a, b0, b1;
#pragma unroll
    for (int e = 0; e < 16; ++e) {
      const int v  = e >> 1, hi = e & 1;
      const int ka = k0 + (g << 4) + e;
      float av = Arow[ka < K ? ka : K - 1];
      a[e] = (ka < K) ? (_Float16)av : (_Float16)0.0f;
      const int kb = k0 + (g << 4) + e;
      _Float16 bv0 = Wr0[kb < K ? kb : K - 1];
      _Float16 bv1 = Wr1[kb < K ? kb : K - 1];
      b0[e] = (kb < K) ? bv0 : (_Float16)0.0f;
      b1[e] = (kb < K) ? bv1 : (_Float16)0.0f;
    }
    c0 = __builtin_amdgcn_wmma_f32_16x16x32_f16(false, a, false, b0, (short)0, c0, false, false);
    c1 = __builtin_amdgcn_wmma_f32_16x16x32_f16(false, a, false, b1, (short)0, c1, false, false);
  }
  // ---- epilogue: bias + SiLU + residual ----
#pragma unroll
  for (int r = 0; r < 8; ++r) {                        // C/D: VGPR r -> row r + 8*g
    const int row = mbase + (g << 3) + r;
    float v0 = c0[r], v1 = c1[r];
    if (bias) { v0 += (n0 < N) ? bias[n0] : 0.0f; v1 += (n1 < N) ? bias[n1] : 0.0f; }
    if (act) {
      v0 = v0 * (1.0f / (1.0f + __expf(-v0)));
      v1 = v1 * (1.0f / (1.0f + __expf(-v1)));
    }
    if (Res) {
      if (n0 < N) v0 += Res[(size_t)row * N + n0];
      if (n1 < N) v1 += Res[(size_t)row * N + n1];
    }
    if (n0 < N) C[(size_t)row * N + n0] = v0;
    if (n1 < N) C[(size_t)row * N + n1] = v1;
  }
}

// ---------------- pair geometry: distances + real spherical harmonics ----------------
__global__ void k_pair(const float* __restrict__ cart, const float* __restrict__ sm,
                       float* __restrict__ dist, float* __restrict__ sph)
{
  int p = blockIdx.x * 256 + threadIdx.x;
  if (p >= NPAIR) return;
  int a = p / (NUMATOM - 1), n = p % (NUMATOM - 1);
  int t = n + (n >= a);
  float isa = 1.0f / sm[a], ist = 1.0f / sm[t];
  float dx = cart[3*t+0]*ist - cart[3*a+0]*isa;
  float dy = cart[3*t+1]*ist - cart[3*a+1]*isa;
  float dz = cart[3*t+2]*ist - cart[3*a+2]*isa;
  float r  = sqrtf(dx*dx + dy*dy + dz*dz);
  dist[p] = r;
  const float c0 = 0.28209479177387814f, c1 = 0.4886025119029199f;
  float inv = 1.0f / r;
  sph[0*NPAIR+p] = c0;
  sph[1*NPAIR+p] = c1 * dy * inv;
  sph[2*NPAIR+p] = c1 * dz * inv;
  sph[3*NPAIR+p] = c1 * dx * inv;
}

// ---------------- radial basis: rad = exp(-(w*(r-mu))^2) ----------------
__global__ void k_rad(const float* __restrict__ emb, const float* __restrict__ dist,
                      float* __restrict__ rad)
{
  int i = blockIdx.x * 256 + threadIdx.x;
  if (i >= NPAIR * NWAVE) return;
  int p = i >> 3, m = i & 7;
  float w  = emb[(size_t)p * EMB_OUT + m];
  float mu = emb[(size_t)p * EMB_OUT + NWAVE + m];
  float x  = w * (dist[p] - mu);
  rad[i] = __expf(-x * x);
}

// ---------------- initial center orbital + density (radial_func[0] lazily) ------
// rad[a*127 ..][8] (4 KB, reused by all 128 threads x 127 iters) is staged to LDS
// via ASYNCcnt-tracked global_load_async_to_lds_b128.
__global__ void k_center0(const float* __restrict__ emb, const float* __restrict__ rad,
                          const float* __restrict__ sph, const float* __restrict__ cc,
                          float* __restrict__ co, float* __restrict__ dens)
{
  int a = blockIdx.x, j = threadIdx.x;          // 128 x 128
  __shared__ float sr[1024];                    // 127*8 = 1016 used
  {
    const float* gsrc = rad + (size_t)a * (NUMATOM - 1) * 8;
    async_g2l_b128(gsrc + j * 4,        &sr[j * 4]);
    async_g2l_b128(gsrc + 512 + j * 4,  &sr[512 + j * 4]);
    async_wait0();
    __syncthreads();
  }
  float ck0[8], ck1[8];
#pragma unroll
  for (int m = 0; m < 8; ++m) {
    ck0[m] = cc[((size_t)(0*NCONTRACT) + j) * 8 + m];
    ck1[m] = cc[((size_t)(1*NCONTRACT) + j) * 8 + m];
  }
  float a0 = 0, a1 = 0, a2 = 0, a3 = 0;
  for (int n = 0; n < NUMATOM - 1; ++n) {
    int p = a * (NUMATOM - 1) + n;
    const float* rp = &sr[n * 8];
    float rf0 = 0, rf1 = 0;
#pragma unroll
    for (int m = 0; m < 8; ++m) { float rm = rp[m]; rf0 += ck0[m]*rm; rf1 += ck1[m]*rm; }
    float w0 = emb[(size_t)p*EMB_OUT + 2*NWAVE + j]             * rf0;
    float w1 = emb[(size_t)p*EMB_OUT + 2*NWAVE + NCONTRACT + j] * rf1;
    a0 += w0 * sph[p];
    a1 += w1 * sph[NPAIR   + p];
    a2 += w1 * sph[2*NPAIR + p];
    a3 += w1 * sph[3*NPAIR + p];
  }
  const float inv = 1.0f / (NUMATOM - 1);
  a0 *= inv; a1 *= inv; a2 *= inv; a3 *= inv;
  co[((size_t)a*4+0)*NCONTRACT+j] = a0;
  co[((size_t)a*4+1)*NCONTRACT+j] = a1;
  co[((size_t)a*4+2)*NCONTRACT+j] = a2;
  co[((size_t)a*4+3)*NCONTRACT+j] = a3;
  dens[((size_t)a*2+0)*NCONTRACT+j] = a0*a0;
  dens[((size_t)a*2+1)*NCONTRACT+j] = a1*a1 + a2*a2 + a3*a3;
}

// ---------------- message-passing interaction (one iteration) ----------------
__global__ void k_mp_inter(const float* __restrict__ rad, const float* __restrict__ sph,
                           const float* __restrict__ cc, const float* __restrict__ cgout,
                           const float* __restrict__ coeff, const float* __restrict__ co_in,
                           float* __restrict__ co_out, float* __restrict__ dens,
                           int level, int cgoff)
{
  CG_TABLES
  int a = blockIdx.x, j = threadIdx.x;
  __shared__ float sr[1024];
  {
    const float* gsrc = rad + (size_t)a * (NUMATOM - 1) * 8;
    async_g2l_b128(gsrc + j * 4,        &sr[j * 4]);
    async_g2l_b128(gsrc + 512 + j * 4,  &sr[512 + j * 4]);
    async_wait0();
    __syncthreads();
  }
  float ck0[8], ck1[8];
#pragma unroll
  for (int m = 0; m < 8; ++m) {
    ck0[m] = cc[(((size_t)level*2+0)*NCONTRACT + j)*8 + m];
    ck1[m] = cc[(((size_t)level*2+1)*NCONTRACT + j)*8 + m];
  }
  float acc[4] = {0, 0, 0, 0};
  for (int n = 0; n < NUMATOM - 1; ++n) {
    int p = a * (NUMATOM - 1) + n;
    int t = n + (n >= a);                          // neighlist[1][p], closed form
    const float* rp = &sr[n * 8];
    float rf0 = 0, rf1 = 0;
#pragma unroll
    for (int m = 0; m < 8; ++m) { float rm = rp[m]; rf0 += ck0[m]*rm; rf1 += ck1[m]*rm; }
    float w0 = rf0 * coeff[((size_t)t*2+0)*NCONTRACT + j];
    float w1 = rf1 * coeff[((size_t)t*2+1)*NCONTRACT + j];
    float orb[4];
    orb[0] = w0 * sph[p];
    orb[1] = w1 * sph[NPAIR   + p];
    orb[2] = w1 * sph[2*NPAIR + p];
    orb[3] = w1 * sph[3*NPAIR + p];
    float io1[4];
#pragma unroll
    for (int s = 0; s < 4; ++s) io1[s] = co_in[((size_t)t*4+s)*NCONTRACT + j];
    float cgl[5];
#pragma unroll
    for (int ch = 0; ch < 5; ++ch) cgl[ch] = cgout[(size_t)p*((NLCH)*3) + cgoff + ch];
#pragma unroll
    for (int q = 0; q < NPATH; ++q)
      acc[IA[q]] += io1[I1[q]] * orb[I2[q]] * cgl[IC[q]] * SGN[q];
  }
  const float inv = 1.0f / (NUMATOM - 1);
  float c0 = acc[0]*inv, c1 = acc[1]*inv, c2 = acc[2]*inv, c3 = acc[3]*inv;
  co_out[((size_t)a*4+0)*NCONTRACT+j] = c0;
  co_out[((size_t)a*4+1)*NCONTRACT+j] = c1;
  co_out[((size_t)a*4+2)*NCONTRACT+j] = c2;
  co_out[((size_t)a*4+3)*NCONTRACT+j] = c3;
  const float is2 = 0.7071067811865476f;
  size_t d0 = ((size_t)a*2+0)*NCONTRACT + j, d1 = ((size_t)a*2+1)*NCONTRACT + j;
  dens[d0] = (dens[d0] + c0*c0) * is2;
  dens[d1] = (dens[d1] + c1*c1 + c2*c2 + c3*c3) * is2;
}

// ---------------- nn-pair density (nn_interaction) ----------------
__global__ void k_nn_density(const float* __restrict__ rad, const float* __restrict__ sph,
                             const float* __restrict__ cc, const float* __restrict__ cgout,
                             const float* __restrict__ co, float* __restrict__ dnn)
{
  CG_TABLES
  int q = blockIdx.x, j = threadIdx.x;
  int a = NELE + q / (NUMATOM - NELE - 1);
  int n = NELE + q % (NUMATOM - NELE - 1);
  int p = a * (NUMATOM - 1) + n;
  int t = n + (n >= a);
  float ck0[8], ck1[8];                       // radial_func[-1] => level 3
#pragma unroll
  for (int m = 0; m < 8; ++m) {
    ck0[m] = cc[(((size_t)3*2+0)*NCONTRACT + j)*8 + m];
    ck1[m] = cc[(((size_t)3*2+1)*NCONTRACT + j)*8 + m];
  }
  const float* rp = rad + (size_t)p * 8;
  float rf0 = 0, rf1 = 0;
#pragma unroll
  for (int m = 0; m < 8; ++m) { float rm = rp[m]; rf0 += ck0[m]*rm; rf1 += ck1[m]*rm; }
  float orb[4];
  orb[0] = rf0 * sph[p];
  orb[1] = rf1 * sph[NPAIR   + p];
  orb[2] = rf1 * sph[2*NPAIR + p];
  orb[3] = rf1 * sph[3*NPAIR + p];
  float io1[4];
#pragma unroll
  for (int s = 0; s < 4; ++s) io1[s] = co[((size_t)t*4+s)*NCONTRACT + j];
  float cgl[5];
#pragma unroll
  for (int ch = 0; ch < 5; ++ch) cgl[ch] = cgout[(size_t)p*((NLCH)*3) + 2*NLCH + ch];
  float acc[4] = {0, 0, 0, 0};
#pragma unroll
  for (int k = 0; k < NPATH; ++k)
    acc[IA[k]] += io1[I1[k]] * orb[I2[k]] * cgl[IC[k]] * SGN[k];
  dnn[((size_t)q*2+0)*NCONTRACT + j] = acc[0]*acc[0];
  dnn[((size_t)q*2+1)*NCONTRACT + j] = acc[1]*acc[1] + acc[2]*acc[2] + acc[3]*acc[3];
}

// ---------------- electron-nucleus decay accumulation ----------------
__global__ void k_decay_en(const float* __restrict__ neout, const float* __restrict__ dist,
                           float* __restrict__ den)
{
  int a = blockIdx.x, tid = threadIdx.x;       // 16 x 512 (i*32+d)
  float acc = 0;
  for (int m = 0; m < NUMATOM - NELE; ++m) {
    int q = a * (NUMATOM - NELE) + m;
    int p = a * (NUMATOM - 1) + (NELE - 1) + m;  // en pairs: n = 15..126
    float r  = dist[p];
    float x0 = neout[(size_t)q*1024 + tid];
    float x1 = neout[(size_t)q*1024 + 512 + tid];
    acc += x0 * __expf(-fabsf(x1 * r));
  }
  den[(size_t)a*512 + tid] = acc;
}

// ---------------- cusp reduction ----------------
__global__ void k_cusp(const float* __restrict__ cusp, const float* __restrict__ dist,
                       float* __restrict__ red)
{
  int p = blockIdx.x * 256 + threadIdx.x;
  float v = 0;
  if (p < NPAIR) { float r = dist[p]; v = cusp[p] * r / (1.0f + r); }
  for (int off = 16; off; off >>= 1) v += __shfl_down(v, off, 32);
  if ((threadIdx.x & 31) == 0) atomicAdd(&red[16], v);
}

// ---------------- 32 x slogdet(16x16) + log|psi| ----------------
__global__ void k_slogdet(const float* __restrict__ nn, const float* __restrict__ de,
                          const float* __restrict__ sp, const float* __restrict__ red,
                          float* __restrict__ out)
{
  __shared__ float Ms[NDET][NELE * NELE];
  __shared__ float sd[NDET], sg[NDET];
  int k = threadIdx.x;                         // one det per thread (32 threads)
  float* Mk = Ms[k];
  for (int i = 0; i < NELE; ++i)
    for (int j = 0; j < NELE; ++j)
      Mk[i*16+j] = nn[i*512 + j*32 + k] * de[i*512 + j*32 + k] * sp[(i*16+j)*32 + k];
  float sign = 1.0f, ld = 0.0f;
  for (int c = 0; c < 16; ++c) {
    int piv = c; float pv = fabsf(Mk[c*16+c]);
    for (int r = c + 1; r < 16; ++r) { float v = fabsf(Mk[r*16+c]); if (v > pv) { pv = v; piv = r; } }
    if (piv != c) {
      for (int j = 0; j < 16; ++j) { float t = Mk[c*16+j]; Mk[c*16+j] = Mk[piv*16+j]; Mk[piv*16+j] = t; }
      sign = -sign;
    }
    float d = Mk[c*16+c];
    if (d < 0.0f) sign = -sign; else if (d == 0.0f) sign = 0.0f;
    ld += logf(fmaxf(fabsf(d), 1e-37f));
    float invd = (d != 0.0f) ? (1.0f / d) : 0.0f;
    for (int r = c + 1; r < 16; ++r) {
      float f = Mk[r*16+c] * invd;
      for (int j = c; j < 16; ++j) Mk[r*16+j] -= f * Mk[c*16+j];
    }
  }
  sd[k] = ld; sg[k] = sign;
  __syncthreads();
  if (k == 0) {
    float mx = -3.4e38f;
    for (int i = 0; i < NDET; ++i) mx = fmaxf(mx, sd[i]);
    float psi = 0;
    for (int i = 0; i < NDET; ++i) psi += sg[i] * __expf(sd[i] - mx);
    float ap = fabsf(psi);
    out[0] = logf(ap > EPSON ? ap : EPSON) + mx + red[16];
  }
}

// ---------------- nn-gaussian reductions ----------------
__global__ void k_nn_reduce(const float* __restrict__ y2, const float* __restrict__ dist,
                            float* __restrict__ red)
{
  int q = blockIdx.x * 256 + threadIdx.x;
  bool ok = q < NNP;
  float r = 1.0f; const float* y = y2;
  if (ok) {
    int a = NELE + q / (NUMATOM - NELE - 1);
    int n = NELE + q % (NUMATOM - NELE - 1);
    r = dist[a * (NUMATOM - 1) + n];
    y = y2 + (size_t)q * (3 * NGAUSS);
  }
  float invr = 1.0f / r;
#pragma unroll
  for (int c = 0; c < 16; ++c) {
    float val = 0;
    if (ok) {
      if (c < NGAUSS) { float t1 = y[c] * r, t2 = y[NGAUSS + c] * invr; val = -(t1*t1 + t2*t2); }
      else            { val = y[NGAUSS + c]; }   // y[16 + (c-8)]
    }
    for (int off = 16; off; off >>= 1) val += __shfl_down(val, off, 32);
    if ((threadIdx.x & 31) == 0) atomicAdd(&red[c], val);
  }
}

__global__ void k_final(const float* __restrict__ red, float* __restrict__ out)
{
  if (threadIdx.x == 0 && blockIdx.x == 0) {
    float mx = red[0];
    for (int g = 1; g < NGAUSS; ++g) mx = fmaxf(mx, red[g]);
    float s = 0;
    for (int g = 0; g < NGAUSS; ++g) s += red[8 + g] * __expf(red[g] - mx);
    float av = fabsf(s);
    out[1] = 2.0f * (mx + logf(av > EPSON ? av : EPSON));
  }
}

// =====================================================================
extern "C" void kernel_launch(void* const* d_in, const int* in_sizes, int n_in,
                              void* d_out, int out_size, void* d_ws, size_t ws_size,
                              hipStream_t stream)
{
  (void)in_sizes; (void)n_in; (void)out_size; (void)ws_size;
  // ---- input map (pytree leaves, dict insertion order; MLP = 12 leaves:
  //      Win, bin, W00,b00, W01,b01, W10,b10, W11,b11, Wout, bout) ----
  const float* cart    = (const float*)d_in[0];
  const float* smass   = (const float*)d_in[1];
  const float* emb_in  = (const float*)d_in[2];
  const float* spin_in = (const float*)d_in[3];
  const float* cusp    = (const float*)d_in[4];
  const float* cc      = (const float*)d_in[5];   // (4,2,128,8)
  const int B_SPIN = 7, B_NEIGH = 19, B_CG = 31, B_MP0 = 43, B_MP1 = 55,
            B_DET = 67, B_NE = 79, B_OUT = 91;
  const int* en_list = (const int*)d_in[104];     // en_neighlist

  // ---- workspace bump allocator ----
  char* base = (char*)d_ws; size_t off = 0;
  auto allocH = [&](size_t n) { _Float16* p = (_Float16*)(base + off);
                                off = (off + n*sizeof(_Float16) + 255) & ~(size_t)255; return p; };
  auto allocF = [&](size_t n) { float* p = (float*)(base + off);
                                off = (off + n*sizeof(float) + 255) & ~(size_t)255; return p; };

  float* dist  = allocF(NPAIR);
  float* sph   = allocF(4 * (size_t)NPAIR);
  float* embc  = allocF((size_t)NPAIR * EMB_OUT);
  float* rad   = allocF((size_t)NPAIR * NWAVE);
  float* cgout = allocF((size_t)NPAIR * 3 * NLCH);
  float* coA   = allocF((size_t)NUMATOM * 4 * NCONTRACT);
  float* coB   = allocF((size_t)NUMATOM * 4 * NCONTRACT);
  float* dens  = allocF((size_t)NUMATOM * 2 * NCONTRACT);
  float* cmp   = allocF((size_t)NUMATOM * 256);
  float* spout = allocF((size_t)NELE * NELE * NDET);
  float* neout = allocF((size_t)ENP * 2 * NELE * NDET);
  float* den   = allocF((size_t)NELE * NELE * NDET);
  float* nnd   = allocF((size_t)NELE * NELE * NDET);
  float* dnn   = allocF((size_t)NNP * 2 * NCONTRACT);
  float* y2    = allocF((size_t)NNP * 3 * NGAUSS);
  float* red   = allocF(32);
  float* a0    = allocF((size_t)NNP * FEAT_MP);
  float* a1    = allocF((size_t)NNP * FEAT_MP);
  float* a2    = allocF((size_t)NNP * FEAT_MP);

  hipMemsetAsync(red, 0, 32 * sizeof(float), stream);

  auto gemm = [&](const float* A, int lda, const int* ridx, const _Float16* Wt,
                  const float* b, const float* Res, float* C, int M, int K, int N, int act) {
    dim3 grid((N + 31) / 32, M / 16), blk(32);
    k_gemm_wmma<<<grid, blk, 0, stream>>>(A, lda, ridx, Wt, b, Res, C, M, K, N, act);
  };

  struct Mlp { _Float16* w[6]; const float* b[6]; int din, feat, nout; };
  auto prep = [&](int bi, int din, int feat, int nout) {
    Mlp m; m.din = din; m.feat = feat; m.nout = nout;
    const int widx[6] = {0, 2, 4, 6, 8, 10}, bidx[6] = {1, 3, 5, 7, 9, 11};
    const int wK[6] = {din, feat, feat, feat, feat, feat};
    const int wN[6] = {feat, feat, feat, feat, feat, nout};
    for (int i = 0; i < 6; ++i) {
      int sz = wK[i] * wN[i];
      m.w[i] = allocH((size_t)sz);
      m.b[i] = (const float*)d_in[bi + bidx[i]];
      k_cvt_tr<<<(sz + 255) / 256, 256, 0, stream>>>((const float*)d_in[bi + widx[i]],
                                                     m.w[i], wK[i], wN[i]);
    }
    return m;
  };
  auto runMlp = [&](const Mlp& m, const float* X, int lda, const int* ridx, int M, float* out) {
    gemm(X,  lda,    ridx,    m.w[0], m.b[0], nullptr, a0,  M, m.din,  m.feat, 1);
    gemm(a0, m.feat, nullptr, m.w[1], m.b[1], nullptr, a1,  M, m.feat, m.feat, 1);
    gemm(a1, m.feat, nullptr, m.w[2], m.b[2], a0,      a2,  M, m.feat, m.feat, 1);
    gemm(a2, m.feat, nullptr, m.w[3], m.b[3], nullptr, a1,  M, m.feat, m.feat, 1);
    gemm(a1, m.feat, nullptr, m.w[4], m.b[4], a2,      a0,  M, m.feat, m.feat, 1);
    gemm(a0, m.feat, nullptr, m.w[5], m.b[5], nullptr, out, M, m.feat, m.nout, 0);
  };

  Mlp mSpin  = prep(B_SPIN,  DSPIN, FEAT_EMB, NDET);
  Mlp mNeigh = prep(B_NEIGH, DEMB,  FEAT_EMB, EMB_OUT);
  Mlp mCg    = prep(B_CG,    DEMB,  FEAT_EMB, 3 * NLCH);
  Mlp mMp0   = prep(B_MP0,   RMAXL * NCONTRACT, FEAT_MP, RMAXL * NCONTRACT);
  Mlp mMp1   = prep(B_MP1,   RMAXL * NCONTRACT, FEAT_MP, RMAXL * NCONTRACT);
  Mlp mDet   = prep(B_DET,   RMAXL * NCONTRACT, FEAT_MP, NELE * NDET);
  Mlp mNe    = prep(B_NE,    DEMB,  FEAT_EMB, 2 * NELE * NDET);
  Mlp mOut   = prep(B_OUT,   RMAXL * NCONTRACT, FEAT_MP, 3 * NGAUSS);

  // ---- forward pipeline ----
  k_pair<<<(NPAIR + 255) / 256, 256, 0, stream>>>(cart, smass, dist, sph);
  runMlp(mNeigh, emb_in, DEMB, nullptr, NPAIR, embc);
  k_rad<<<(NPAIR * NWAVE + 255) / 256, 256, 0, stream>>>(embc, dist, rad);
  runMlp(mCg, emb_in, DEMB, nullptr, NPAIR, cgout);
  k_center0<<<NUMATOM, NCONTRACT, 0, stream>>>(embc, rad, sph, cc, coA, dens);

  runMlp(mMp0, dens, RMAXL * NCONTRACT, nullptr, NUMATOM, cmp);
  k_mp_inter<<<NUMATOM, NCONTRACT, 0, stream>>>(rad, sph, cc, cgout, cmp, coA, coB, dens, 1, 0);
  runMlp(mMp1, dens, RMAXL * NCONTRACT, nullptr, NUMATOM, cmp);
  k_mp_inter<<<NUMATOM, NCONTRACT, 0, stream>>>(rad, sph, cc, cgout, cmp, coB, coA, dens, 2, NLCH);

  runMlp(mSpin, spin_in, DSPIN, nullptr, NELE * NELE, spout);
  runMlp(mNe, emb_in, DEMB, en_list, ENP, neout);
  k_decay_en<<<NELE, NELE * NDET, 0, stream>>>(neout, dist, den);
  runMlp(mDet, dens, RMAXL * NCONTRACT, nullptr, NELE, nnd);
  k_cusp<<<(NPAIR + 255) / 256, 256, 0, stream>>>(cusp, dist, red);
  k_slogdet<<<1, NDET, 0, stream>>>(nnd, den, spout, red, (float*)d_out);

  k_nn_density<<<NNP, NCONTRACT, 0, stream>>>(rad, sph, cc, cgout, coA, dnn);
  runMlp(mOut, dnn, RMAXL * NCONTRACT, nullptr, NNP, y2);
  k_nn_reduce<<<(NNP + 255) / 256, 256, 0, stream>>>(y2, dist, red);
  k_final<<<1, 1, 0, stream>>>(red, (float*)d_out);
}
